// SynergyGAT_48155173322904
// MI455X (gfx1250) — matrix operand order repaired and checked by reading, not compiled, not run on hardware
//
#include <hip/hip_runtime.h>

typedef __attribute__((ext_vector_type(16))) _Float16 v16h;
typedef __attribute__((ext_vector_type(8)))  _Float16 v8h;
typedef __attribute__((ext_vector_type(8)))  float    v8f;

#define NEG_SLOPE 0.2f

// ---------------------------------------------------------------------------
// helpers
// ---------------------------------------------------------------------------
__device__ __forceinline__ float lrelu(float v) {
    return v > 0.0f ? v : NEG_SLOPE * v;
}
__device__ __forceinline__ float elu(float v) {
    return v > 0.0f ? v : (__expf(v) - 1.0f);
}
// float atomic max via int/uint monotonicity trick (works for mixed signs)
__device__ __forceinline__ void atomicMaxF(float* addr, float val) {
    if (val >= 0.0f) atomicMax((int*)addr, __float_as_int(val));
    else             atomicMin((unsigned int*)addr, __float_as_uint(val));
}
__device__ __forceinline__ void edge_sd(const int* __restrict__ ei, int E, int e,
                                        int& s, int& d) {
    if (e < E) { s = ei[e]; d = ei[E + e]; }
    else       { s = e - E; d = e - E; }          // self loops appended
}

// ---------------------------------------------------------------------------
// x (f32, [N,64]) -> padded f16 [Npad,64]
// ---------------------------------------------------------------------------
__global__ void k_convert_x(const float* __restrict__ x, _Float16* __restrict__ A16,
                            int N, int total) {
    int i = blockIdx.x * blockDim.x + threadIdx.x;
    if (i >= total) return;
    int n = i >> 6;                               // /64
    float v = (n < N) ? x[i] : 0.0f;
    A16[i] = (_Float16)v;
}

// ---------------------------------------------------------------------------
// pack W [K, Cout] (f32 row-major) into WMMA B-fragment order (f16).
// Fragment layout used here (16-bit B, 32x16; contiguous k, same as A): lane L holds col = L%16,
// element j (0..15) holds K = (L/16)*16 + j.
// Bp index = ((tn*kBlocks + kb)*32 + L)*16 + j
// ---------------------------------------------------------------------------
__global__ void k_pack_w(const float* __restrict__ W, _Float16* __restrict__ Bp,
                         int kBlocks, int Cout, int total) {
    int i = blockIdx.x * blockDim.x + threadIdx.x;
    if (i >= total) return;
    int j    = i & 15;
    int L    = (i >> 4) & 31;
    int rest = i >> 9;
    int kb   = rest % kBlocks;
    int tn   = rest / kBlocks;
    int k = kb * 32 + ((L >> 4) << 4) + j;
    int n = tn * 16 + (L & 15);
    Bp[i] = (_Float16)W[(size_t)k * Cout + n];
}

// ---------------------------------------------------------------------------
// GEMM: C[Npad, Cout] = A[Npad, K](f16) x Bp(packed f16), f32 accumulate.
// One wave computes a 16x64 tile: 4 accumulators share one A fragment per
// k-step (4x A reuse), v_wmma_f32_16x16x32_f16 over K/32 steps.
// 128-thread blocks = 4 waves, each owning one 16x64 tile.
// ---------------------------------------------------------------------------
__global__ __launch_bounds__(128) void k_gemm(const _Float16* __restrict__ A,
                                              const _Float16* __restrict__ Bp,
                                              float* __restrict__ C,
                                              int tilesM, int tilesN4,
                                              int kBlocks, int K, int Cout) {
    int gtile = blockIdx.x * 4 + (threadIdx.x >> 5);
    if (gtile >= tilesM * tilesN4) return;        // wave-uniform guard
    int tm  = gtile % tilesM;
    int tn4 = gtile / tilesM;
    int lane = threadIdx.x & 31;
    int c16  = lane & 15;
    int hs   = lane >> 4;                         // lane-group select

    const _Float16* arow = A + (size_t)(tm * 16 + c16) * K;
    const size_t bstride = (size_t)kBlocks * 32 * 16;   // halves per 16-col block
    v8f acc0 = {}, acc1 = {}, acc2 = {}, acc3 = {};

    for (int kb = 0; kb < kBlocks; ++kb) {
        // A fragment: lane holds row M=c16; 16 elems K=kb*32+hs*16..+15
        const _Float16* ap = arow + kb * 32 + hs * 16;
        v16h a = *reinterpret_cast<const v16h*>(ap);
        // 4 B fragments: contiguous 32B loads from the packed buffer
        const _Float16* bbase =
            Bp + ((size_t)((tn4 * 4) * kBlocks + kb) * 32 + lane) * 16;
        v16h b0 = *reinterpret_cast<const v16h*>(bbase);
        v16h b1 = *reinterpret_cast<const v16h*>(bbase + bstride);
        v16h b2 = *reinterpret_cast<const v16h*>(bbase + 2 * bstride);
        v16h b3 = *reinterpret_cast<const v16h*>(bbase + 3 * bstride);
        acc0 = __builtin_amdgcn_wmma_f32_16x16x32_f16(false, a, false, b0, (short)0, acc0, false, false);
        acc1 = __builtin_amdgcn_wmma_f32_16x16x32_f16(false, a, false, b1, (short)0, acc1, false, false);
        acc2 = __builtin_amdgcn_wmma_f32_16x16x32_f16(false, a, false, b2, (short)0, acc2, false, false);
        acc3 = __builtin_amdgcn_wmma_f32_16x16x32_f16(false, a, false, b3, (short)0, acc3, false, false);
    }
    // D layout: VGPR v -> row = tm*16 + hs*8 + v, col block u -> tn4*64 + u*16 + c16
    float* cp = C + (size_t)(tm * 16 + hs * 8) * Cout + (size_t)tn4 * 64 + c16;
#pragma unroll
    for (int v = 0; v < 8; ++v) {
        cp[(size_t)v * Cout +  0] = acc0[v];
        cp[(size_t)v * Cout + 16] = acc1[v];
        cp[(size_t)v * Cout + 32] = acc2[v];
        cp[(size_t)v * Cout + 48] = acc3[v];
    }
}

// ---------------------------------------------------------------------------
// alpha_s[n,h] = dot(h[n,h,:], a_src[h,:]) ; same for alpha_d
// ---------------------------------------------------------------------------
__global__ void k_alpha(const float* __restrict__ h, const float* __restrict__ a_src,
                        const float* __restrict__ a_dst, float* __restrict__ as_,
                        float* __restrict__ ad_, int H, int Cc, int total) {
    int i = blockIdx.x * blockDim.x + threadIdx.x;
    if (i >= total) return;
    int n = i / H, hd = i % H;
    const float* hp  = h + (size_t)n * H * Cc + (size_t)hd * Cc;
    const float* asp = a_src + (size_t)hd * Cc;
    const float* adp = a_dst + (size_t)hd * Cc;
    float s0 = 0.0f, s1 = 0.0f;
    for (int c = 0; c < Cc; ++c) { float v = hp[c]; s0 += v * asp[c]; s1 += v * adp[c]; }
    as_[i] = s0; ad_[i] = s1;
}

__global__ void k_init_ms(float* __restrict__ m, float* __restrict__ s, int total) {
    int i = blockIdx.x * blockDim.x + threadIdx.x;
    if (i >= total) return;
    m[i] = -3.0e38f;
    s[i] = 0.0f;
}

__global__ void k_zero(float* __restrict__ p, size_t total) {
    size_t i = (size_t)blockIdx.x * blockDim.x + threadIdx.x;
    if (i >= total) return;
    p[i] = 0.0f;
}

// segment max over incoming edges (thread per (edge, head))
__global__ void k_edge_max(const int* __restrict__ ei, int E,
                           const float* __restrict__ as_, const float* __restrict__ ad_,
                           float* __restrict__ m, int H, int total) {
    int i = blockIdx.x * blockDim.x + threadIdx.x;
    if (i >= total) return;
    int e = i / H, hd = i % H;
    int s, d; edge_sd(ei, E, e, s, d);
    float v = lrelu(as_[s * H + hd] + ad_[d * H + hd]);
    atomicMaxF(&m[d * H + hd], v);
}

// ex = exp(e - m[dst]); segment sum; stash ex per edge
__global__ void k_edge_expsum(const int* __restrict__ ei, int E,
                              const float* __restrict__ as_, const float* __restrict__ ad_,
                              const float* __restrict__ m, float* __restrict__ sden,
                              float* __restrict__ exbuf, int H, int total) {
    int i = blockIdx.x * blockDim.x + threadIdx.x;
    if (i >= total) return;
    int e = i / H, hd = i % H;
    int s, d; edge_sd(ei, E, e, s, d);
    float v  = lrelu(as_[s * H + hd] + ad_[d * H + hd]);
    float ex = __expf(v - m[d * H + hd]);
    exbuf[i] = ex;
    atomicAdd(&sden[d * H + hd], ex);
}

// normalize: exbuf[e,h] /= sden[dst,h]  (one divide per edge-head, not per channel)
__global__ void k_alpha_norm(const int* __restrict__ ei, int E,
                             const float* __restrict__ sden, float* __restrict__ exbuf,
                             int H, int total) {
    int i = blockIdx.x * blockDim.x + threadIdx.x;
    if (i >= total) return;
    int e = i / H, hd = i % H;
    int s, d; edge_sd(ei, E, e, s, d);
    (void)s;
    exbuf[i] = exbuf[i] / sden[d * H + hd];
}

// weighted scatter-add: agg[dst] += h[src] * alpha   (block per edge, float4 gather)
__global__ void k_edge_agg(const int* __restrict__ ei, int E,
                           const float* __restrict__ h,
                           const float* __restrict__ alpha,
                           float* __restrict__ agg, int H, int Cc) {
    int e = blockIdx.x;
    int s, d; edge_sd(ei, E, e, s, d);
    int HC = H * Cc;
    const float* hp = h + (size_t)s * HC;
    float* op = agg + (size_t)d * HC;
    for (int c = threadIdx.x * 4; c < HC; c += blockDim.x * 4) {
        int hd = c / Cc;
        float al = alpha[(size_t)e * H + hd];
        float4 hv = *reinterpret_cast<const float4*>(hp + c);
        atomicAdd(&op[c + 0], hv.x * al);
        atomicAdd(&op[c + 1], hv.y * al);
        atomicAdd(&op[c + 2], hv.z * al);
        atomicAdd(&op[c + 3], hv.w * al);
    }
}

// bias + ELU -> f16 activations (zero padded rows) for the next WMMA GEMM
__global__ void k_post_f16(const float* __restrict__ agg, const float* __restrict__ b,
                           _Float16* __restrict__ A16, int N, int HC, size_t total) {
    size_t i = (size_t)blockIdx.x * blockDim.x + threadIdx.x;
    if (i >= total) return;
    int n = (int)(i / HC);
    int c = (int)(i % HC);
    float v = 0.0f;
    if (n < N) v = elu(agg[i] + b[c]);
    A16[i] = (_Float16)v;
}

// bias + ELU -> f32 (last GAT layer, feeds the MLP head)
__global__ void k_post_f32(const float* __restrict__ agg, const float* __restrict__ b,
                           float* __restrict__ outp, size_t total, int HC) {
    size_t i = (size_t)blockIdx.x * blockDim.x + threadIdx.x;
    if (i >= total) return;
    int c = (int)(i % HC);
    outp[i] = elu(agg[i] + b[c]);
}

// MLP head: sigmoid(relu(h@fc1 + b1) @ fc2 + b2), block of 64 threads per node
__global__ __launch_bounds__(64) void k_mlp(const float* __restrict__ h3,
                                            const float* __restrict__ w1,
                                            const float* __restrict__ b1,
                                            const float* __restrict__ w2,
                                            const float* __restrict__ b2,
                                            float* __restrict__ out, int N) {
    __shared__ float xs[128];
    __shared__ float hs[64];
    int n = blockIdx.x;
    int t = threadIdx.x;
    xs[t]      = h3[(size_t)n * 128 + t];
    xs[t + 64] = h3[(size_t)n * 128 + t + 64];
    __syncthreads();
    float acc = b1[t];
    for (int k = 0; k < 128; ++k) acc += xs[k] * w1[k * 64 + t];
    acc = fmaxf(acc, 0.0f);
    hs[t] = acc * w2[t];
    __syncthreads();
    if (t < 32) hs[t] += hs[t + 32];
    __syncthreads();
    if (t == 0) {
        float r = 0.0f;
        for (int k = 0; k < 32; ++k) r += hs[k];
        r += b2[0];
        out[n] = 1.0f / (1.0f + __expf(-r));
    }
}

// ---------------------------------------------------------------------------
extern "C" void kernel_launch(void* const* d_in, const int* in_sizes, int n_in,
                              void* d_out, int out_size, void* d_ws, size_t ws_size,
                              hipStream_t stream) {
    const float* x = (const float*)d_in[0];
    const int*  ei = (const int*)d_in[1];
    const int E    = in_sizes[1] / 2;          // 400000
    const int N    = in_sizes[0] / 64;         // 50000
    const int Npad = (N + 15) & ~15;           // 50000
    const int ET   = E + N;                    // with self loops

    char* ws = (char*)d_ws;
    size_t o = 0;
    auto take = [&](size_t bytes) -> char* {
        char* p = ws + o;
        o += (bytes + 255) & ~(size_t)255;
        return p;
    };
    _Float16* A16  = (_Float16*)take((size_t)Npad * 512 * 2);   // GEMM input (padded)
    _Float16* Bp   = (_Float16*)take((size_t)512 * 512 * 2);    // packed weight
    float*    h    = (float*)take((size_t)Npad * 512 * 4);      // GEMM output
    float*    agg  = (float*)take((size_t)Npad * 512 * 4);      // scatter accumulator
    float*    as_  = (float*)take((size_t)Npad * 4 * 4);
    float*    ad_  = (float*)take((size_t)Npad * 4 * 4);
    float*    mbuf = (float*)take((size_t)Npad * 4 * 4);
    float*    sden = (float*)take((size_t)Npad * 4 * 4);
    float*    exb  = (float*)take((size_t)ET * 4 * 4);

    { // x -> f16 padded
        int tot = Npad * 64;
        k_convert_x<<<(tot + 255) / 256, 256, 0, stream>>>(x, A16, N, tot);
    }

    struct LayerP { int K, H, C, wi, asi, adi, bi; };
    const LayerP Ls[3] = {
        { 64, 4, 128,  2,  3,  4,  5 },
        { 512, 4, 128, 6,  7,  8,  9 },
        { 512, 1, 128, 10, 11, 12, 13 },
    };

    for (int li = 0; li < 3; ++li) {
        const LayerP& P = Ls[li];
        const int HC = P.H * P.C;
        const int kBlocks = P.K / 32;
        const float* W = (const float*)d_in[P.wi];

        { int tot = P.K * HC;
          k_pack_w<<<(tot + 255) / 256, 256, 0, stream>>>(W, Bp, kBlocks, HC, tot); }

        { int tilesM = Npad / 16, tilesN4 = HC / 64;
          int totTiles = tilesM * tilesN4;
          k_gemm<<<(totTiles + 3) / 4, 128, 0, stream>>>(A16, Bp, h, tilesM, tilesN4,
                                                         kBlocks, P.K, HC); }

        { int tot = N * P.H;
          k_alpha<<<(tot + 255) / 256, 256, 0, stream>>>(
              h, (const float*)d_in[P.asi], (const float*)d_in[P.adi],
              as_, ad_, P.H, P.C, tot); }

        { int tot = N * P.H;
          k_init_ms<<<(tot + 255) / 256, 256, 0, stream>>>(mbuf, sden, tot); }

        { int tot = ET * P.H;
          k_edge_max<<<(tot + 255) / 256, 256, 0, stream>>>(ei, E, as_, ad_, mbuf, P.H, tot); }

        { int tot = ET * P.H;
          k_edge_expsum<<<(tot + 255) / 256, 256, 0, stream>>>(
              ei, E, as_, ad_, mbuf, sden, exb, P.H, tot); }

        { int tot = ET * P.H;
          k_alpha_norm<<<(tot + 255) / 256, 256, 0, stream>>>(ei, E, sden, exb, P.H, tot); }

        { size_t tot = (size_t)N * HC;
          k_zero<<<(unsigned)((tot + 255) / 256), 256, 0, stream>>>(agg, tot); }

        k_edge_agg<<<ET, HC / 4, 0, stream>>>(ei, E, h, exb, agg, P.H, P.C);

        const float* b = (const float*)d_in[P.bi];
        if (li < 2) {
            size_t tot = (size_t)Npad * HC;
            k_post_f16<<<(unsigned)((tot + 255) / 256), 256, 0, stream>>>(agg, b, A16, N, HC, tot);
        } else {
            size_t tot = (size_t)N * HC;   // reuse h as f32 activation buffer
            k_post_f32<<<(unsigned)((tot + 255) / 256), 256, 0, stream>>>(agg, b, h, tot, HC);
        }
    }

    k_mlp<<<N, 64, 0, stream>>>(h, (const float*)d_in[14], (const float*)d_in[15],
                                (const float*)d_in[16], (const float*)d_in[17],
                                (float*)d_out, N);
}
